// SimplifiedDynamicModel_86887188398505
// MI455X (gfx1250) — matrix operand order repaired and checked by reading, not compiled, not run on hardware
//
#include <hip/hip_runtime.h>

typedef __bf16 bhalf;
typedef __attribute__((ext_vector_type(16))) __bf16 v16bf;
typedef __attribute__((ext_vector_type(8)))  __bf16 v8bf;
typedef __attribute__((ext_vector_type(8)))  float  v8f;
typedef __attribute__((ext_vector_type(4)))  unsigned int v4u;
typedef __attribute__((ext_vector_type(8)))  int v8i;
typedef __attribute__((ext_vector_type(4)))  int v4i;

#define NNODE 100

__device__ __forceinline__ bhalf f2bf(float f) {
    unsigned u = __float_as_uint(f);
    u += 0x7FFFu + ((u >> 16) & 1u);          // round-to-nearest-even
    unsigned short h = (unsigned short)(u >> 16);
    bhalf r; __builtin_memcpy(&r, &h, 2); return r;
}
__device__ __forceinline__ float bf2f(bhalf b) {
    unsigned short h; __builtin_memcpy(&h, &b, 2);
    return __uint_as_float(((unsigned)h) << 16);
}

// ---------------------------------------------------------------------------
// Node embedding: x = node_feats[M,3] @ W_node[3,256] + b_node  -> fp32 + bf16
// ---------------------------------------------------------------------------
__global__ __launch_bounds__(256)
void k_embed(const float* __restrict__ nf, const float* __restrict__ W,
             const float* __restrict__ bias, float* __restrict__ xf,
             bhalf* __restrict__ xb)
{
    int i = blockIdx.x * 256 + threadIdx.x;   // over M*256
    int row = i >> 8, c = i & 255;
    float v = nf[row*3+0]*W[c] + nf[row*3+1]*W[256+c] + nf[row*3+2]*W[512+c] + bias[c];
    xf[i] = v;
    xb[i] = f2bf(v);
}

// ---------------------------------------------------------------------------
// Weight transpose+convert: W[K,N] fp32 -> Wt[N,K] bf16 (B operand layout)
// ---------------------------------------------------------------------------
__global__ __launch_bounds__(256)
void k_transpose(const float* __restrict__ W, bhalf* __restrict__ Wt, int K, int N)
{
    int i = blockIdx.x * 256 + threadIdx.x;
    if (i >= K * N) return;
    int k = i / N, n = i - k * N;
    Wt[(size_t)n * K + k] = f2bf(W[(size_t)k * N + n]);
}

// ---------------------------------------------------------------------------
// WMMA GEMM: Out[M,N] = A_bf16[M,K] @ Bt_bf16[N,K]^T + bias
// 256 threads = 8 waves; block tile 128x64; wave tile 32x32 (2x2 WMMA).
// Double-buffered LDS (one barrier per 32-K step), compile-time epilogue.
// ---------------------------------------------------------------------------
#define LDT 40   // padded LDS stride (halves): 80B rows, 16B aligned, no conflicts

template<int RELU, int OBF16>
__global__ __launch_bounds__(256)
void k_gemm_bf16(const bhalf* __restrict__ A, const bhalf* __restrict__ Bt,
                 const float* __restrict__ bias, void* __restrict__ Out,
                 int M, int N, int K)
{
    __shared__ bhalf As[2][128 * LDT];   // 2 x 10.0 KB
    __shared__ bhalf Bs[2][64 * LDT];    // 2 x  5.0 KB

    const int tid  = threadIdx.x;
    const int wave = tid >> 5, lane = tid & 31;
    const int wm = wave & 3, wn = wave >> 2;           // 4x2 wave grid
    const int M0 = blockIdx.x * 128, N0 = blockIdx.y * 64;

    const int arow = tid >> 1, acol = (tid & 1) << 4;  // A: 128 rows x 32 halves
    const int brow = tid >> 2, bcol = (tid & 3) << 3;  // B:  64 rows x 32 halves
    const int hl  = lane & 15;
    const int kbA = (lane < 16) ? 0 : 8;               // ISA 16-bit A layout
    const int koB = (lane < 16) ? 0 : 8;               // ISA 16-bit B layout

    const bhalf* gA = A  + (size_t)(M0 + arow) * K + acol;
    const bhalf* gB = Bt + (size_t)(N0 + brow) * K + bcol;

    v8f acc[2][2] = {};

    // Prologue: stage tile 0 into buffer 0.
    v8bf ra0 = *(const v8bf*)(gA);
    v8bf ra1 = *(const v8bf*)(gA + 8);
    v8bf rb0 = *(const v8bf*)(gB);
    *(v8bf*)(&As[0][arow*LDT + acol])     = ra0;
    *(v8bf*)(&As[0][arow*LDT + acol + 8]) = ra1;
    *(v8bf*)(&Bs[0][brow*LDT + bcol])     = rb0;
    __syncthreads();

    int p = 0;
    for (int k0 = 0; k0 < K; k0 += 32, p ^= 1) {
        const bool next = (k0 + 32) < K;
        if (next) {                           // long-latency fetch first
            ra0 = *(const v8bf*)(gA + k0 + 32);
            ra1 = *(const v8bf*)(gA + k0 + 32 + 8);
            rb0 = *(const v8bf*)(gB + k0 + 32);
        }

        v16bf af[2], bfr[2];
#pragma unroll
        for (int t = 0; t < 2; ++t) {
            const bhalf* pa = &As[p][(wm*32 + t*16 + hl) * LDT + kbA];
            v8bf alo = *(const v8bf*)pa;
            v8bf ahi = *(const v8bf*)(pa + 16);
            af[t] = __builtin_shufflevector(alo, ahi, 0,1,2,3,4,5,6,7,8,9,10,11,12,13,14,15);
            const bhalf* pb = &Bs[p][(wn*32 + t*16 + hl) * LDT + koB];
            v8bf blo = *(const v8bf*)pb;
            v8bf bhi = *(const v8bf*)(pb + 16);
            bfr[t] = __builtin_shufflevector(blo, bhi, 0,1,2,3,4,5,6,7,8,9,10,11,12,13,14,15);
        }
#pragma unroll
        for (int tm = 0; tm < 2; ++tm)
#pragma unroll
            for (int tn = 0; tn < 2; ++tn)
                acc[tm][tn] = __builtin_amdgcn_wmma_f32_16x16x32_bf16(
                    false, af[tm], false, bfr[tn], (short)0, acc[tm][tn], false, false);

        if (next) {                           // stage next tile into other buffer
            *(v8bf*)(&As[p^1][arow*LDT + acol])     = ra0;
            *(v8bf*)(&As[p^1][arow*LDT + acol + 8]) = ra1;
            *(v8bf*)(&Bs[p^1][brow*LDT + bcol])     = rb0;
            __syncthreads();
        }
    }

    // Branchless epilogue (flags are compile-time).
#pragma unroll
    for (int tm = 0; tm < 2; ++tm) {
#pragma unroll
        for (int tn = 0; tn < 2; ++tn) {
            int n  = N0 + wn*32 + tn*16 + hl;
            int mb = M0 + wm*32 + tm*16 + ((lane >> 4) << 3);
            float bv = bias[n];
            if (OBF16) {
                bhalf* op = (bhalf*)Out + (size_t)mb * N + n;
#pragma unroll
                for (int v = 0; v < 8; ++v) {
                    float val = acc[tm][tn][v] + bv;
                    if (RELU) val = fmaxf(val, 0.0f);
                    op[(size_t)v * N] = f2bf(val);
                }
            } else {
                float* op = (float*)Out + (size_t)mb * N + n;
#pragma unroll
                for (int v = 0; v < 8; ++v) {
                    float val = acc[tm][tn][v] + bv;
                    if (RELU) val = fmaxf(val, 0.0f);
                    op[(size_t)v * N] = val;
                }
            }
        }
    }
}

// ---------------------------------------------------------------------------
// Attention: per (batch, head), flash-style online softmax, dh=32, N=100
// qkv layout [B, N, 768] bf16 : q=[0,256) k=[256,512) v=[512,768)
// ---------------------------------------------------------------------------
__global__ __launch_bounds__(128)
void k_attn(const bhalf* __restrict__ qkv, bhalf* __restrict__ attn)
{
    __shared__ float Kh[NNODE * 32];
    __shared__ float Vh[NNODE * 32];
    int b = blockIdx.x >> 3, h = blockIdx.x & 7;
    const bhalf* base = qkv + (size_t)b * NNODE * 768;

    for (int idx = threadIdx.x; idx < NNODE * 32; idx += 128) {
        int j = idx >> 5, d = idx & 31;
        Kh[idx] = bf2f(base[(size_t)j * 768 + 256 + h*32 + d]);
        Vh[idx] = bf2f(base[(size_t)j * 768 + 512 + h*32 + d]);
    }
    __syncthreads();

    int q = threadIdx.x;
    if (q < NNODE) {
        float Qr[32];
#pragma unroll
        for (int d = 0; d < 32; ++d) Qr[d] = bf2f(base[(size_t)q * 768 + h*32 + d]);
        const float scale = 0.17677669529663687f;   // 1/sqrt(32)
        float m = -3.0e38f, l = 0.0f, acc[32];
#pragma unroll
        for (int d = 0; d < 32; ++d) acc[d] = 0.0f;
        for (int j = 0; j < NNODE; ++j) {
            float s = 0.0f;
#pragma unroll
            for (int d = 0; d < 32; ++d) s += Qr[d] * Kh[j*32 + d];
            s *= scale;
            float nm = fmaxf(m, s);
            float c = __expf(m - nm);
            float e = __expf(s - nm);
            l = l * c + e;
#pragma unroll
            for (int d = 0; d < 32; ++d) acc[d] = acc[d] * c + e * Vh[j*32 + d];
            m = nm;
        }
        float inv = 1.0f / l;
        bhalf* op = attn + (size_t)(b * NNODE + q) * 256 + h*32;
#pragma unroll
        for (int d = 0; d < 32; ++d) op[d] = f2bf(acc[d] * inv);
    }
}

// ---------------------------------------------------------------------------
// Residual + LayerNorm: x = LN(x + y)*g + b -> fp32 + bf16. One wave per row.
// ---------------------------------------------------------------------------
__global__ __launch_bounds__(256)
void k_add_ln(const float* __restrict__ x, const float* __restrict__ y,
              const float* __restrict__ g, const float* __restrict__ bb,
              float* __restrict__ xo, bhalf* __restrict__ xb)
{
    int wave = threadIdx.x >> 5, lane = threadIdx.x & 31;
    size_t row = (size_t)blockIdx.x * 8 + wave;
    const float* xr = x + row * 256;
    const float* yr = y + row * 256;
    float v[8], s = 0.f, ss = 0.f;
#pragma unroll
    for (int k = 0; k < 8; ++k) {
        int c = lane + 32*k;
        v[k] = xr[c] + yr[c];
        s += v[k]; ss += v[k] * v[k];
    }
#pragma unroll
    for (int off = 16; off > 0; off >>= 1) {
        s  += __shfl_xor(s,  off, 32);
        ss += __shfl_xor(ss, off, 32);
    }
    float mean = s * (1.0f/256.0f);
    float var  = ss * (1.0f/256.0f) - mean * mean;
    float rstd = rsqrtf(var + 1e-5f);
#pragma unroll
    for (int k = 0; k < 8; ++k) {
        int c = lane + 32*k;
        float o = (v[k] - mean) * rstd * g[c] + bb[c];
        xo[row*256 + c] = o;
        xb[row*256 + c] = f2bf(o);
    }
}

// ---------------------------------------------------------------------------
// Pairwise scores: S[b,i,j] = x[b,i] . x[b,j]   (T_SOFTMAX = 1)
// Embedding slab (100x256 bf16 = 51200 B, contiguous) is pulled into LDS with
// one Tensor-Data-Mover descriptor issued by wave 0 (TENSORcnt-tracked).
// ---------------------------------------------------------------------------
__global__ __launch_bounds__(256)
void k_scores(const bhalf* __restrict__ xb, float* __restrict__ S)
{
    __shared__ bhalf xs[NNODE * 256];   // 50 KB
    int b = blockIdx.x;
    const bhalf* src = xb + (size_t)b * NNODE * 256;

#if __has_builtin(__builtin_amdgcn_tensor_load_to_lds) && __has_builtin(__builtin_amdgcn_s_wait_tensorcnt)
    if (threadIdx.x < 32) {             // one wave issues the DMA
        unsigned long long ga = (unsigned long long)(uintptr_t)src;
        unsigned lds = (unsigned)(uintptr_t)(void*)xs;
        const unsigned ELEMS = NNODE * 256u;         // 25600 x 2B
        v4u g0; v8i g1; v4i g2 = {0,0,0,0}, g3 = {0,0,0,0};
        v8i g4 = {0,0,0,0,0,0,0,0};
        // D# group0: count=1 | lds_addr | global_addr[56:0] | type=2
        g0[0] = 1u;
        g0[1] = lds;
        g0[2] = (unsigned)(ga & 0xffffffffu);
        g0[3] = (unsigned)((ga >> 32) & 0x01ffffffu) | (2u << 30);
        // D# group1: data_size=2B; tensor_dim0=tile_dim0=ELEMS; dim1=1
        g1[0] = (int)(1u << 16);                     // data_size=1 -> 2 bytes
        g1[1] = (int)((ELEMS & 0xffffu) << 16);      // tensor_dim0[15:0] @ bits 63:48
        g1[2] = (int)((ELEMS >> 16) | (1u << 16));   // tensor_dim0[31:16] | tensor_dim1=1
        g1[3] = (int)((ELEMS & 0xffffu) << 16);      // tile_dim0 = ELEMS
        g1[4] = 1;                                   // tile_dim1=1, tile_dim2=0
        g1[5] = (int)ELEMS;                          // tensor_dim0_stride
        g1[6] = 0;
        g1[7] = 0;
        __builtin_amdgcn_tensor_load_to_lds(g0, g1, g2, g3, g4, 0);
        __builtin_amdgcn_s_wait_tensorcnt((short)0);
    }
    __syncthreads();
#else
    for (int idx = threadIdx.x; idx < NNODE * 256; idx += 256) xs[idx] = src[idx];
    __syncthreads();
#endif

    for (int e = threadIdx.x; e < NNODE * NNODE; e += 256) {
        int i = e / NNODE, j = e - i * NNODE;
        float acc = 0.f;
#pragma unroll 8
        for (int d = 0; d < 256; ++d)
            acc += bf2f(xs[i*256 + d]) * bf2f(xs[j*256 + d]);
        S[(size_t)b * NNODE * NNODE + e] = acc;
    }
}

// ---------------------------------------------------------------------------
// Sequential routing decode: one wave32 per batch, shfl reductions only.
// out: route[B,200] (as float) then log_probs[B]
// ---------------------------------------------------------------------------
__global__ __launch_bounds__(32)
void k_route(const float* __restrict__ S, const float* __restrict__ demand,
             const float* __restrict__ capacity, const float* __restrict__ gumbel,
             float* __restrict__ out, int B, int steps)
{
    int b = blockIdx.x, lane = threadIdx.x;
    float dem[4]; int vis[4];
#pragma unroll
    for (int t = 0; t < 4; ++t) {
        int n = lane + 32*t;
        dem[t] = (n < NNODE) ? demand[b*NNODE + n] : 0.f;
        vis[t] = (n == 0) ? 1 : ((n < NNODE) ? 0 : 1);
    }
    float cap = capacity[b], lp = 0.f;
    int cur = 0;
    if (lane == 0) out[(size_t)b * 200] = 0.f;

    for (int step = 0; step < steps; ++step) {
        const float* Srow = S + ((size_t)b * NNODE + cur) * NNODE;
        const float* grow = gumbel + ((size_t)step * B + b) * NNODE;
        float sc[4];
        float bestv = -3.0e38f; int besti = 0; float mx = -1e9f; int anyf = 0;
#pragma unroll
        for (int t = 0; t < 4; ++t) {
            int n = lane + 32*t;
            if (n < NNODE) {
                int f = (!vis[t]) && (dem[t] <= cap);
                anyf |= f;
                sc[t] = f ? Srow[n] : -1e9f;
                float v = sc[t] + grow[n];
                if (v > bestv) { bestv = v; besti = n; }
                mx = fmaxf(mx, sc[t]);
            } else sc[t] = -1e9f;
        }
#pragma unroll
        for (int off = 16; off > 0; off >>= 1) {
            float ov = __shfl_xor(bestv, off, 32);
            int   oi = __shfl_xor(besti, off, 32);
            if (ov > bestv || (ov == bestv && oi < besti)) { bestv = ov; besti = oi; }
            mx = fmaxf(mx, __shfl_xor(mx, off, 32));
        }
        int active = __any(anyf);
        float se = 0.f;
#pragma unroll
        for (int t = 0; t < 4; ++t) {
            int n = lane + 32*t;
            if (n < NNODE) se += __expf(sc[t] - mx);
        }
#pragma unroll
        for (int off = 16; off > 0; off >>= 1) se += __shfl_xor(se, off, 32);

        int choice = besti, node = 0;
        if (active) {
            float scc = Srow[choice];               // choice is feasible -> masked==score
            lp += scc - mx - __logf(se);
            int tc = choice >> 5, lc = choice & 31;
            float dl = dem[0];
            if (tc == 1) dl = dem[1];
            if (tc == 2) dl = dem[2];
            if (tc == 3) dl = dem[3];
            cap -= __shfl(dl, lc, 32);
            if (lane == lc) {
                if (tc == 0) vis[0] = 1;
                if (tc == 1) vis[1] = 1;
                if (tc == 2) vis[2] = 1;
                if (tc == 3) vis[3] = 1;
            }
            cur = choice;
            node = choice;
        }
        if (lane == 0) out[(size_t)b * 200 + step + 1] = (float)node;
    }
    if (lane == 0) out[(size_t)B * 200 + b] = lp;
}

// ---------------------------------------------------------------------------
extern "C" void kernel_launch(void* const* d_in, const int* in_sizes, int n_in,
                              void* d_out, int out_size, void* d_ws, size_t ws_size,
                              hipStream_t stream)
{
    const float* node_feats = (const float*)d_in[0];
    const float* demand     = (const float*)d_in[1];
    const float* capacity   = (const float*)d_in[2];
    const float* gumbel     = (const float*)d_in[3];
    const float* W_node     = (const float*)d_in[4];
    const float* b_node     = (const float*)d_in[5];
    const float* qkv_w      = (const float*)d_in[6];
    const float* qkv_b      = (const float*)d_in[7];
    const float* out_w      = (const float*)d_in[8];
    const float* out_b      = (const float*)d_in[9];
    const float* ln1_s      = (const float*)d_in[10];
    const float* ln1_b      = (const float*)d_in[11];
    const float* w1         = (const float*)d_in[12];
    const float* b1         = (const float*)d_in[13];
    const float* w2         = (const float*)d_in[14];
    const float* b2         = (const float*)d_in[15];
    const float* ln2_s      = (const float*)d_in[16];
    const float* ln2_b      = (const float*)d_in[17];
    (void)in_sizes; (void)n_in; (void)out_size; (void)ws_size;

    const int B = 512, Nn = 100, H = 256, L = 3, F = 512;
    const int M = B * Nn;                        // 51200 = 400 * 128

    char* ws = (char*)d_ws;
    size_t off = 0;
    auto alloc = [&](size_t bytes) -> void* {
        void* p = ws + off;
        off += (bytes + 255) & ~(size_t)255;
        return p;
    };
    float* x_f32  = (float*)alloc((size_t)M * H * 4);        // 52.4 MB
    bhalf* x_b    = (bhalf*)alloc((size_t)M * H * 2);        // 26.2 MB
    bhalf* qkv_o  = (bhalf*)alloc((size_t)M * 3 * H * 2);    // 78.6 MB
    bhalf* bufA   = (bhalf*)alloc((size_t)M * F * 2);        // 52.4 MB (attn out / ffn hidden)
    float* bufB   = (float*)alloc((size_t)M * H * 4);        // 52.4 MB (proj / ffn out)
    float* S      = (float*)alloc((size_t)B * Nn * Nn * 4);  // 20.5 MB
    bhalf* qkv_t  = (bhalf*)alloc((size_t)L * 3 * H * H * 2);
    bhalf* out_t  = (bhalf*)alloc((size_t)L * H * H * 2);
    bhalf* w1_t   = (bhalf*)alloc((size_t)L * H * F * 2);
    bhalf* w2_t   = (bhalf*)alloc((size_t)L * F * H * 2);

    // Pre-transpose + convert weights to bf16 [N,K] for the WMMA B layout.
    for (int l = 0; l < L; ++l) {
        k_transpose<<<(H*3*H + 255)/256, 256, 0, stream>>>(qkv_w + (size_t)l*H*3*H, qkv_t + (size_t)l*3*H*H, H, 3*H);
        k_transpose<<<(H*H   + 255)/256, 256, 0, stream>>>(out_w + (size_t)l*H*H,   out_t + (size_t)l*H*H,   H, H);
        k_transpose<<<(H*F   + 255)/256, 256, 0, stream>>>(w1    + (size_t)l*H*F,   w1_t  + (size_t)l*F*H,   H, F);
        k_transpose<<<(F*H   + 255)/256, 256, 0, stream>>>(w2    + (size_t)l*F*H,   w2_t  + (size_t)l*H*F,   F, H);
    }

    k_embed<<<M, 256, 0, stream>>>(node_feats, W_node, b_node, x_f32, x_b);

    for (int l = 0; l < L; ++l) {
        dim3 gqkv(M/128, (3*H)/64);
        k_gemm_bf16<0,1><<<gqkv, 256, 0, stream>>>(x_b, qkv_t + (size_t)l*3*H*H,
                                                   qkv_b + (size_t)l*3*H, qkv_o, M, 3*H, H);
        k_attn<<<B*8, 128, 0, stream>>>(qkv_o, bufA);
        dim3 gout(M/128, H/64);
        k_gemm_bf16<0,0><<<gout, 256, 0, stream>>>(bufA, out_t + (size_t)l*H*H,
                                                   out_b + (size_t)l*H, bufB, M, H, H);
        k_add_ln<<<M/8, 256, 0, stream>>>(x_f32, bufB, ln1_s + l*H, ln1_b + l*H, x_f32, x_b);
        dim3 gf1(M/128, F/64);
        k_gemm_bf16<1,1><<<gf1, 256, 0, stream>>>(x_b, w1_t + (size_t)l*H*F,
                                                  b1 + (size_t)l*F, bufA, M, F, H);
        dim3 gf2(M/128, H/64);
        k_gemm_bf16<0,0><<<gf2, 256, 0, stream>>>(bufA, w2_t + (size_t)l*F*H,
                                                  b2 + (size_t)l*H, bufB, M, H, F);
        k_add_ln<<<M/8, 256, 0, stream>>>(x_f32, bufB, ln2_s + l*H, ln2_b + l*H, x_f32, x_b);
    }

    // Precompute pairwise scores once, then run the cheap sequential decode.
    k_scores<<<B, 256, 0, stream>>>(x_b, S);
    k_route<<<B, 32, 0, stream>>>(S, demand, capacity, gumbel, (float*)d_out, B, 199);
}
